// MemoryModule_20212116095157
// MI455X (gfx1250) — matrix operand order repaired and checked by reading, not compiled, not run on hardware
//
#include <hip/hip_runtime.h>
#include <hip/hip_bf16.h>
#include <math.h>

// ---------------------------------------------------------------------------
// MI455X (gfx1250) implementation of the MemoryModule reference.
// Wave32, WMMA bf16 16x16x32 for all GEMMs, TDM (tensor_load_to_lds) double-
// buffered staging of the keys matrix (with TDM LDS row padding to kill bank
// conflicts), fused top-k/softmax/gather, fused gate-MLP epilogue.
// ---------------------------------------------------------------------------

typedef __bf16 bf16;
typedef __bf16 bf16x16 __attribute__((ext_vector_type(16)));
typedef __bf16 bf16x8  __attribute__((ext_vector_type(8)));
typedef float  f32x8   __attribute__((ext_vector_type(8)));
typedef unsigned int u32x4 __attribute__((ext_vector_type(4)));
typedef int  i32x8 __attribute__((ext_vector_type(8)));
typedef int  i32x4 __attribute__((ext_vector_type(4)));

#define NTOK   16384      // 8 * 2048 tokens
#define DIM    1024
#define NSLOT  4096
#define TOPK   8
#define HID    512        // D/2
#define K2     2048       // 2*D

// sims kernel tiling
#define SIMS_THREADS 256
#define SIMS_WAVES   8
#define GRP_SLOTS    128              // slots per group = SIMS_WAVES * 16
#define KTILE        128              // k-extent of one TDM tile
#define NKT          (DIM / KTILE)    // 8
#define NGROUP       (NSLOT / GRP_SLOTS)   // 32
#define TOT_TILES    (NGROUP * NKT)        // 256

// LDS row pads (16 bytes) -> row strides hit distinct banks
#define QS_LD   (DIM + 8)             // bf16 elements (2064 B rows)
#define KT_LD   (KTILE + 8)           // bf16 elements (272 B rows)
#define SB_LD   (GRP_SLOTS + 4)       // f32 elements
#define G_LD    (HID + 4)             // f32 elements

// ---- WMMA helpers ---------------------------------------------------------

__device__ __forceinline__ f32x8 wmma_bf16(bf16x16 a, bf16x16 b, f32x8 c) {
  // v_wmma_f32_16x16x32_bf16
  return __builtin_amdgcn_wmma_f32_16x16x32_bf16(
      /*neg_a=*/false, a, /*neg_b=*/false, b,
      /*c_mod=*/(short)0, c, /*reuse_a=*/false, /*reuse_b=*/false);
}

// A fragment: 16x32 (MxK) bf16.  Lane L holds row M=L&15,
// K = kb + (L>>4)*16 .. +15 (contiguous, matches B fragment).
__device__ __forceinline__ bf16x16 load_a_frag(const bf16* base, int ld,
                                               int kb, int lane) {
  const int m  = lane & 15;
  const int k0 = kb + ((lane >> 4) << 4);
  const bf16* p = base + (long)m * ld + k0;
  bf16x8 lo = *(const bf16x8*)(p);
  bf16x8 hi = *(const bf16x8*)(p + 8);
  return __builtin_shufflevector(lo, hi, 0,1,2,3,4,5,6,7,8,9,10,11,12,13,14,15);
}

// B fragment: 32x16 (KxN) bf16 from a row-major [N,K] matrix (B(k,n)=M[n,k],
// contiguous in k per lane).  Lane L holds column N=L&15, K = kb+(L>>4)*16..+15.
__device__ __forceinline__ bf16x16 load_b_frag(const bf16* base, int ld,
                                               int n0, int kb, int lane) {
  const int n  = n0 + (lane & 15);
  const int k0 = kb + ((lane >> 4) << 4);
  return *(const bf16x16*)(base + (long)n * ld + k0);
}

// ---- Tensor Data Mover: async 2-D bf16 tile load (global -> LDS) ----------
// D# per ISA 8.3/8.4: group0 = {count=1, lds_addr, global_addr, type=2};
// group1 = {data_size=2B, pad ctl, tensor dims, tile dims, dim0 stride};
// groups 2/3 zero (2-D tensor).  LDS padding: pad_interval code 5 = 64 DWORDs
// (one 256 B tile row), pad_amount code 3 = 4 DWORDs (16 B) -> 272 B rows.
// Issued by one full wave (TDM ignores EXEC); tracked with TENSORcnt.
__device__ __forceinline__ void tdm_load_tile_bf16(
    unsigned lds_addr, const bf16* gaddr,
    int tile_k, int tile_rows, int tensor_k, int tensor_rows, int row_stride,
    int pad_en, int pad_interval_code, int pad_amount_code) {
  unsigned long long ga = (unsigned long long)(const void*)gaddr;
  u32x4 g0;
  g0[0] = 1u;                                        // count=1 (valid, user)
  g0[1] = lds_addr;                                  // LDS byte address
  g0[2] = (unsigned)(ga & 0xFFFFFFFFu);              // global_addr[31:0]
  g0[3] = (unsigned)((ga >> 32) & 0x1FFFFFFu) | (2u << 30);  // addr[56:32]|type=2
  i32x8 g1;
  g1[0] = (1 << 16)                                  // data_size=1 -> 2 bytes
        | (pad_en << 20)                             // pad_enable
        | (pad_interval_code << 22)                  // pad_interval
        | (pad_amount_code << 25);                   // pad_amount
  g1[1] = (tensor_k & 0xFFFF) << 16;                 // tensor_dim0[15:0] @bit48
  g1[2] = ((unsigned)tensor_k >> 16) | ((tensor_rows & 0xFFFF) << 16);
  g1[3] = ((unsigned)tensor_rows >> 16) | ((tile_k & 0xFFFF) << 16); // tile_dim0
  g1[4] = (tile_rows & 0xFFFF);                      // tile_dim1 (tile_dim2=0)
  g1[5] = row_stride;                                // tensor_dim0_stride[31:0]
  g1[6] = 0;                                         // stride hi / dim1_stride
  g1[7] = 0;
  i32x4 z = {0, 0, 0, 0};
#if defined(__clang_major__) && (__clang_major__ >= 23)
  i32x8 zz = {0, 0, 0, 0, 0, 0, 0, 0};
  __builtin_amdgcn_tensor_load_to_lds(g0, g1, z, z, zz, 0);
#else
  __builtin_amdgcn_tensor_load_to_lds(g0, g1, z, z, 0);
#endif
}

// ---- small utility kernels ------------------------------------------------

__global__ void mm_f32_to_bf16(const float* __restrict__ src,
                               bf16* __restrict__ dst, long n) {
  long i = (long)blockIdx.x * blockDim.x + threadIdx.x;
  long stride = (long)gridDim.x * blockDim.x;
  for (; i < n; i += stride) dst[i] = (bf16)src[i];
}

// keys -> l2-normalized bf16 rows
__global__ void mm_l2norm_f32_to_bf16(const float* __restrict__ src,
                                      bf16* __restrict__ dst, int cols) {
  __shared__ float red[256];
  const long row = blockIdx.x;
  const float* s = src + row * cols;
  float ss = 0.f;
  for (int c = threadIdx.x; c < cols; c += 256) { float v = s[c]; ss += v * v; }
  red[threadIdx.x] = ss;
  __syncthreads();
  for (int off = 128; off > 0; off >>= 1) {
    if (threadIdx.x < off) red[threadIdx.x] += red[threadIdx.x + off];
    __syncthreads();
  }
  const float inv = 1.0f / fmaxf(sqrtf(red[0]), 1e-12f);
  for (int c = threadIdx.x; c < cols; c += 256)
    dst[row * cols + c] = (bf16)(s[c] * inv);
}

// in-place l2 normalize bf16 rows (queries)
__global__ void mm_l2norm_bf16_inplace(bf16* __restrict__ data, int cols) {
  __shared__ float red[256];
  const long row = blockIdx.x;
  bf16* p = data + row * cols;
  float ss = 0.f;
  for (int c = threadIdx.x; c < cols; c += 256) { float v = (float)p[c]; ss += v * v; }
  red[threadIdx.x] = ss;
  __syncthreads();
  for (int off = 128; off > 0; off >>= 1) {
    if (threadIdx.x < off) red[threadIdx.x] += red[threadIdx.x + off];
    __syncthreads();
  }
  const float inv = 1.0f / fmaxf(sqrtf(red[0]), 1e-12f);
  for (int c = threadIdx.x; c < cols; c += 256)
    p[c] = (bf16)((float)p[c] * inv);
}

// ---- GEMM: C[M,Ncols] = A[M,K] * B[Ncols,K]^T, all bf16, f32 accumulate ----
// Block = 128 threads (4 waves); each wave owns a 32x16 output (two 16x16
// accumulators sharing one B fragment -> 6 loads per 2 WMMAs).  grid.y covers
// 64 output columns per block.  Weights stay L2-resident.
__global__ void mm_gemm_bf16_nt(const bf16* __restrict__ A,
                                const bf16* __restrict__ B,
                                bf16* __restrict__ C, int K, int ldc) {
  const int lane = threadIdx.x & 31;
  const int wave = threadIdx.x >> 5;
  const long mBase = (long)blockIdx.x * 32;
  const int  nBase = blockIdx.y * 64 + wave * 16;
  const bf16* Arow0 = A + mBase * K;
  const bf16* Arow1 = Arow0 + (long)16 * K;

  f32x8 acc0 = {}, acc1 = {};
  for (int kb = 0; kb < K; kb += 32) {
    bf16x16 b  = load_b_frag(B, K, nBase, kb, lane);
    bf16x16 a0 = load_a_frag(Arow0, K, kb, lane);
    bf16x16 a1 = load_a_frag(Arow1, K, kb, lane);
    acc0 = wmma_bf16(a0, b, acc0);
    acc1 = wmma_bf16(a1, b, acc1);
  }
  const int col    = nBase + (lane & 15);
  const int rowOff = (lane >> 4) << 3;
#pragma unroll
  for (int r = 0; r < 8; ++r) {
    C[(mBase + rowOff + r) * ldc + col]      = (bf16)acc0[r];
    C[(mBase + 16 + rowOff + r) * ldc + col] = (bf16)acc1[r];
  }
}

// ---- fused sims -> top-8 -> softmax -> weighted value gather --------------
// One block = 16 tokens x all 4096 slots.  Query rows staged once in LDS
// (padded rows).  Keys streamed as 128-slot x 128-k bf16 tiles via the Tensor
// Data Mover into a double-buffered LDS tile with TDM row padding; wave 0
// issues the DMA and waits TENSORcnt, the workgroup barrier publishes the
// tile; 8 waves each own one 16x16 sub-tile, 4 WMMAs per barrier.
__global__ void mm_sims_topk_retrieve(const bf16* __restrict__ qn,
                                      const bf16* __restrict__ kn,
                                      const float* __restrict__ values,
                                      bf16* __restrict__ ret_bf) {
  __shared__ __align__(16) bf16 qs[16][QS_LD];               // ~32 KB
  __shared__ __align__(16) bf16 ktile[2][GRP_SLOTS][KT_LD];  // 2 x 34 KB
  __shared__ float simsb[16][SB_LD];                         // ~8 KB
  __shared__ float wbuf[16][TOPK];
  __shared__ int   ibuf[16][TOPK];

  const int tid  = threadIdx.x;
  const int lane = tid & 31;
  const int wave = tid >> 5;
  const long tokBase = (long)blockIdx.x * 16;

  // stage 16x1024 bf16 query rows -> padded LDS rows (128 x 16B per row)
  {
    const uint4* src = (const uint4*)(qn + tokBase * DIM);
    for (int i = tid; i < 16 * (DIM / 8); i += SIMS_THREADS) {
      const int row = i >> 7;          // DIM/8 == 128 chunks per row
      const int col = i & 127;
      *((uint4*)&qs[row][0] + col) = src[(long)row * 128 + col];
    }
  }

  // prologue: wave 0 issues the TDM load of tile 0 into buffer 0
  if (wave == 0) {
    tdm_load_tile_bf16((unsigned)(unsigned long long)&ktile[0][0][0],
                       kn, KTILE, GRP_SLOTS, DIM, NSLOT, DIM,
                       /*pad*/1, /*interval 64DW*/5, /*amount 4DW*/3);
  }
  __syncthreads();

  float tv[TOPK];
  int   ti[TOPK];
#pragma unroll
  for (int k = 0; k < TOPK; ++k) { tv[k] = -1e30f; ti[k] = 0; }

  f32x8 acc = {};
  for (int t = 0; t < TOT_TILES; ++t) {
    // current tile must be resident before anyone reads it
    if (wave == 0) __builtin_amdgcn_s_wait_tensorcnt(0);
    __syncthreads();

    // kick off the next tile into the other buffer (overlaps with compute)
    if (wave == 0 && (t + 1) < TOT_TILES) {
      const int tn = t + 1;
      const int gN  = tn >> 3;                  // slot group of next tile
      const int kkN = (tn & (NKT - 1)) * KTILE;
      tdm_load_tile_bf16((unsigned)(unsigned long long)&ktile[(tn & 1)][0][0],
                         kn + (long)gN * GRP_SLOTS * DIM + kkN,
                         KTILE, GRP_SLOTS, DIM, NSLOT, DIM, 1, 5, 3);
    }

    // four WMMA k-steps out of the resident tile (ds_load_b128 operand feed)
    const int kk = (t & (NKT - 1)) * KTILE;
    const bf16* bt = &ktile[t & 1][0][0];
#pragma unroll
    for (int ks = 0; ks < KTILE; ks += 32) {
      bf16x16 a = load_a_frag(&qs[0][0], QS_LD, kk + ks, lane);
      bf16x16 b = load_b_frag(bt, KT_LD, wave * 16, ks, lane);
      acc = wmma_bf16(a, b, acc);
    }

    // end of a slot group: publish sims, fold into per-token top-8
    if ((t & (NKT - 1)) == (NKT - 1)) {
      const int g = t >> 3;
      const int col    = wave * 16 + (lane & 15);
      const int rowOff = (lane >> 4) << 3;
#pragma unroll
      for (int r = 0; r < 8; ++r) simsb[rowOff + r][col] = acc[r];
      acc = f32x8{};
      __syncthreads();

      if (tid < 16) {
        for (int c = 0; c < GRP_SLOTS; ++c) {
          const float v = simsb[tid][c];
          if (v > tv[TOPK - 1]) {
            int p = TOPK - 1;
#pragma unroll
            for (int q = TOPK - 2; q >= 0; --q)
              if (v > tv[q]) p = q;
            for (int q = TOPK - 1; q > p; --q) { tv[q] = tv[q - 1]; ti[q] = ti[q - 1]; }
            tv[p] = v; ti[p] = g * GRP_SLOTS + c;
          }
        }
      }
      __syncthreads();
    }
  }

  // softmax over the 8 survivors (tv[0] is the max: list is sorted desc)
  if (tid < 16) {
    const float mx = tv[0];
    float e[TOPK], s = 0.f;
#pragma unroll
    for (int k = 0; k < TOPK; ++k) { e[k] = expf(tv[k] - mx); s += e[k]; }
    const float inv = 1.0f / s;
#pragma unroll
    for (int k = 0; k < TOPK; ++k) { wbuf[tid][k] = e[k] * inv; ibuf[tid][k] = ti[k]; }
  }
  __syncthreads();

  // retrieved[t,:] = sum_k w_k * values[idx_k,:]   (values resident in L2)
  for (int t = 0; t < 16; ++t) {
    float w[TOPK]; const float* vp[TOPK];
#pragma unroll
    for (int k = 0; k < TOPK; ++k) {
      w[k]  = wbuf[t][k];
      vp[k] = values + (long)ibuf[t][k] * DIM;
    }
    for (int e = tid; e < DIM; e += SIMS_THREADS) {
      float a = 0.f;
#pragma unroll
      for (int k = 0; k < TOPK; ++k) a += w[k] * vp[k][e];
      ret_bf[(tokBase + t) * DIM + e] = (bf16)a;
    }
  }
}

// ---- fused gate MLP + output ---------------------------------------------
// g = gelu([x,proj] @ gW1^T + b1)  (WMMA, K=2048 split across x_bf/proj_bf;
// each wave runs pairs of N-tiles sharing one A fragment -> 3 loads/WMMA)
// gate = sigmoid(g . gW2 + b2);   out = x + gate * proj
__global__ void mm_mlp_gate_out(const float* __restrict__ x,
                                const bf16* __restrict__ x_bf,
                                const bf16* __restrict__ proj_bf,
                                const bf16* __restrict__ gW1_bf,
                                const float* __restrict__ gb1,
                                const float* __restrict__ gW2,
                                const float* __restrict__ gb2,
                                float* __restrict__ out) {
  __shared__ float g[16][G_LD];                      // ~32 KB (padded rows)
  __shared__ float gacc[16];
  __shared__ float gate[16];

  const int tid  = threadIdx.x;
  const int lane = tid & 31;
  const int wave = tid >> 5;
  const long tokBase = (long)blockIdx.x * 16;

  if (tid < 16) gacc[tid] = 0.f;

  for (int pair = 0; pair < 4; ++pair) {
    const int j0 = wave * 8 + pair * 2;              // two adjacent N-tiles
    const int h0 = j0 * 16, h1 = h0 + 16;
    f32x8 acc0 = {}, acc1 = {};
    for (int kb = 0; kb < K2; kb += 32) {
      const bf16* Asrc = (kb < DIM) ? (x_bf + tokBase * DIM + kb)
                                    : (proj_bf + tokBase * DIM + (kb - DIM));
      bf16x16 a  = load_a_frag(Asrc, DIM, 0, lane);
      bf16x16 b0 = load_b_frag(gW1_bf, K2, h0, kb, lane);
      bf16x16 b1 = load_b_frag(gW1_bf, K2, h1, kb, lane);
      acc0 = wmma_bf16(a, b0, acc0);
      acc1 = wmma_bf16(a, b1, acc1);
    }
    const int n      = lane & 15;
    const int rowOff = (lane >> 4) << 3;
    const float bias0 = gb1[h0 + n], bias1 = gb1[h1 + n];
#pragma unroll
    for (int r = 0; r < 8; ++r) {
      const float v0 = acc0[r] + bias0;
      const float v1 = acc1[r] + bias1;
      g[rowOff + r][h0 + n] = 0.5f * v0 * (1.0f + erff(v0 * 0.70710678118f));
      g[rowOff + r][h1 + n] = 0.5f * v1 * (1.0f + erff(v1 * 0.70710678118f));
    }
  }
  __syncthreads();

  {  // gate dot-product: 8 lanes per token over 512 hidden units
    const int t = tid >> 3, part = tid & 7;
    float s = 0.f;
    for (int h = part * 64; h < part * 64 + 64; ++h) s += g[t][h] * gW2[h];
    atomicAdd(&gacc[t], s);
  }
  __syncthreads();
  if (tid < 16) gate[tid] = 1.0f / (1.0f + expf(-(gacc[tid] + gb2[0])));
  __syncthreads();

  for (int t = 0; t < 16; ++t) {
    const float gt = gate[t];
    const long row = (tokBase + t) * DIM;
    for (int e = tid; e < DIM; e += 128)
      out[row + e] = x[row + e] + gt * (float)proj_bf[row + e];
  }
}

// ---------------------------------------------------------------------------

extern "C" void kernel_launch(void* const* d_in, const int* in_sizes, int n_in,
                              void* d_out, int out_size, void* d_ws, size_t ws_size,
                              hipStream_t stream) {
  const float* x      = (const float*)d_in[0];   // [8,2048,1024]
  const float* keys   = (const float*)d_in[1];   // [4096,1024]
  const float* values = (const float*)d_in[2];   // [4096,1024]
  const float* Wq     = (const float*)d_in[3];   // [1024,1024]
  const float* Wo     = (const float*)d_in[4];   // [1024,1024]
  const float* gW1    = (const float*)d_in[5];   // [512,2048]
  const float* gb1    = (const float*)d_in[6];   // [512]
  const float* gW2    = (const float*)d_in[7];   // [1,512]
  const float* gb2    = (const float*)d_in[8];   // [1]
  float* out = (float*)d_out;

  // workspace carve-up (bf16 staging, 256B aligned regions)
  char* w = (char*)d_ws;
  size_t off = 0;
  auto carve = [&](size_t bytes) -> char* {
    char* p = w + off;
    off += (bytes + 255) & ~(size_t)255;
    return p;
  };
  bf16* x_bf    = (bf16*)carve((size_t)NTOK * DIM * 2);   // 32 MB
  bf16* Wq_bf   = (bf16*)carve((size_t)DIM * DIM * 2);    //  2 MB
  bf16* Wo_bf   = (bf16*)carve((size_t)DIM * DIM * 2);    //  2 MB
  bf16* gW1_bf  = (bf16*)carve((size_t)HID * K2 * 2);     //  2 MB
  bf16* kn_bf   = (bf16*)carve((size_t)NSLOT * DIM * 2);  //  8 MB (L2-resident)
  bf16* q_bf    = (bf16*)carve((size_t)NTOK * DIM * 2);   // 32 MB
  bf16* ret_bf  = (bf16*)carve((size_t)NTOK * DIM * 2);   // 32 MB
  bf16* proj_bf = (bf16*)carve((size_t)NTOK * DIM * 2);   // 32 MB
  (void)ws_size; (void)n_in; (void)in_sizes; (void)out_size;

  // 1) bf16 staging of activations and weights
  mm_f32_to_bf16<<<4096, 256, 0, stream>>>(x,   x_bf,   (long)NTOK * DIM);
  mm_f32_to_bf16<<<1024, 256, 0, stream>>>(Wq,  Wq_bf,  (long)DIM * DIM);
  mm_f32_to_bf16<<<1024, 256, 0, stream>>>(Wo,  Wo_bf,  (long)DIM * DIM);
  mm_f32_to_bf16<<<1024, 256, 0, stream>>>(gW1, gW1_bf, (long)HID * K2);

  // 2) normalized keys
  mm_l2norm_f32_to_bf16<<<NSLOT, 256, 0, stream>>>(keys, kn_bf, DIM);

  // 3) queries = x @ Wq^T  (WMMA), then row l2-normalize in place
  mm_gemm_bf16_nt<<<dim3(NTOK / 32, DIM / 64), 128, 0, stream>>>(
      x_bf, Wq_bf, q_bf, DIM, DIM);
  mm_l2norm_bf16_inplace<<<NTOK, 256, 0, stream>>>(q_bf, DIM);

  // 4) fused sims / top-8 / softmax / value gather (TDM-staged keys)
  mm_sims_topk_retrieve<<<NTOK / 16, SIMS_THREADS, 0, stream>>>(
      q_bf, kn_bf, values, ret_bf);

  // 5) proj = retrieved @ Wo^T (WMMA)
  mm_gemm_bf16_nt<<<dim3(NTOK / 32, DIM / 64), 128, 0, stream>>>(
      ret_bf, Wo_bf, proj_bf, DIM, DIM);

  // 6) fused gate MLP + residual output
  mm_mlp_gate_out<<<NTOK / 16, 128, 0, stream>>>(
      x, x_bf, proj_bf, gW1_bf, gb1, gW2, gb2, out);
}
